// MultiHeadSelfAttention_89550068121752
// MI455X (gfx1250) — matrix operand order repaired and checked by reading, not compiled, not run on hardware
//
#include <hip/hip_runtime.h>

#define BB 2
#define SS 2048
#define DD 1024
#define HH 16
#define DHH 64
#define NX (BB * SS * DD)   // 4194304
#define NW (DD * DD)        // 1048576

typedef __attribute__((ext_vector_type(16))) __bf16 v16bf;
typedef __attribute__((ext_vector_type(8)))  __bf16 v8bf;
typedef __attribute__((ext_vector_type(8)))  float  v8f;

__device__ __forceinline__ __bf16 f32_to_bf16(float f) {
  union { float f; unsigned u; } x; x.f = f;
  unsigned r = x.u + 0x7FFFu + ((x.u >> 16) & 1u);
  union { unsigned short s; __bf16 b; } y;
  y.s = (unsigned short)(r >> 16);
  return y.b;
}

// A-fragment (16x32 bf16): lane L (0-15) = row L, K = kb..kb+7 (regs0-3) and
// kb+16..kb+23 (regs4-7), with kb = (L>>4)*8. Two contiguous 16B loads.
__device__ __forceinline__ v16bf load_afrag(const __bf16* p) {
  v8bf lo = *(const v8bf*)p;
  v8bf hi = *(const v8bf*)(p + 16);
  return __builtin_shufflevector(lo, hi, 0,1,2,3,4,5,6,7,8,9,10,11,12,13,14,15);
}

__device__ __forceinline__ v8f wmma_bf16(v16bf a, v16bf b, v8f c) {
  return __builtin_amdgcn_wmma_f32_16x16x32_bf16(
      /*neg_a=*/false, a, /*neg_b=*/false, b,
      /*c_mod=*/(short)0, c, /*reuse_a=*/false, /*reuse_b=*/false);
}

// ---------------------------------------------------------------- converters
__global__ void cvt_f32_bf16(const float* __restrict__ s, __bf16* __restrict__ d, int n) {
  int i = (blockIdx.x * blockDim.x + threadIdx.x) * 4;
  if (i + 3 < n) {
    float4 f = *(const float4*)(s + i);
    d[i + 0] = f32_to_bf16(f.x);
    d[i + 1] = f32_to_bf16(f.y);
    d[i + 2] = f32_to_bf16(f.z);
    d[i + 3] = f32_to_bf16(f.w);
  }
}

// -------------------------------------------------------------- QKV proj GEMM
// [4096 x 1024] x [1024 x 1024] per z-slice (0=Q,1=K,2=V).
// Each wave computes a 32x64 output tile (2 M sub-tiles share each B-frag).
// Q,V stored [B,H,S,DH]; K stored transposed [B,H,DH,S].
__global__ void qkv_gemm(const __bf16* __restrict__ xb, const __bf16* __restrict__ wb,
                         const float* __restrict__ bq, const float* __restrict__ bk,
                         const float* __restrict__ bv,
                         __bf16* __restrict__ qo, __bf16* __restrict__ ko,
                         __bf16* __restrict__ vo) {
  const int which = blockIdx.z;
  const __bf16* W = wb + (size_t)which * NW;
  const float* bias = (which == 0) ? bq : ((which == 1) ? bk : bv);
  const int lane = threadIdx.x & 31;
  const int wave = threadIdx.x >> 5;
  const int m0 = blockIdx.x * 128 + wave * 32;
  const int n0 = blockIdx.y * 64;
  const int row = lane & 15;
  const int kb  = (lane >> 4) * 8;

  v8f acc[2][4] = {{v8f{}, v8f{}, v8f{}, v8f{}}, {v8f{}, v8f{}, v8f{}, v8f{}}};

  for (int k0 = 0; k0 < DD; k0 += 32) {
    v16bf a0 = load_afrag(xb + (size_t)(m0 + row) * DD + k0 + kb);
    v16bf a1 = load_afrag(xb + (size_t)(m0 + 16 + row) * DD + k0 + kb);
    const __bf16* bp = W + (size_t)(k0 + kb) * DD + n0 + (lane & 15);  // lane = N-col of B, K gathered
    __builtin_prefetch(bp + 32 * DD, 0, 3);                 // next K-tile, WGP-near
#pragma unroll
    for (int f = 0; f < 4; ++f) {
      v16bf bfrag;
#pragma unroll
      for (int s = 0; s < 16; ++s) bfrag[s] = bp[(size_t)(s + (s >> 3) * 8) * DD + f * 16];
      acc[0][f] = wmma_bf16(a0, bfrag, acc[0][f]);
      acc[1][f] = wmma_bf16(a1, bfrag, acc[1][f]);
    }
  }

#pragma unroll
  for (int t = 0; t < 2; ++t) {
#pragma unroll
    for (int f = 0; f < 4; ++f) {
      int n = n0 + f * 16 + (lane & 15);
      int h = n >> 6, d = n & 63;
      float bia = bias[n];
#pragma unroll
      for (int r = 0; r < 8; ++r) {
        int m = m0 + t * 16 + r + ((lane >> 4) << 3);
        int b = m >> 11;            // m / 2048
        int s = m & 2047;
        float val = acc[t][f][r] + bia;
        size_t bh = (size_t)(b * HH + h);
        if (which == 1) {
          ko[(bh * DHH + d) * SS + s] = f32_to_bf16(val);   // K transposed
        } else {
          __bf16* dst = (which == 0) ? qo : vo;
          dst[(bh * SS + s) * DHH + d] = f32_to_bf16(val);
        }
      }
    }
  }
}

// ----------------------------------------------------------- flash attention
// block = 4 waves x 16 queries; streams keys in chunks of 64 with online softmax.
__global__ void attn_kernel(const __bf16* __restrict__ q, const __bf16* __restrict__ kT,
                            const __bf16* __restrict__ v, __bf16* __restrict__ ctxo) {
  const int lane = threadIdx.x & 31;
  const int wave = threadIdx.x >> 5;
  const int bh = blockIdx.y;                 // 0..31  (b*H+h)
  const int q0 = blockIdx.x * 64 + wave * 16;
  const int row = lane & 15;
  const int kb  = (lane >> 4) * 8;

  __shared__ __align__(32) __bf16 plds[4][16 * 64];   // 8 KB: P tile per wave

  // Q fragments for d-chunks [0,32) and [32,64)
  const __bf16* qp = q + ((size_t)bh * SS + q0 + row) * DHH + kb;
  v16bf aq0 = load_afrag(qp);
  v16bf aq1 = load_afrag(qp + 32);

  v8f acc[4] = {v8f{}, v8f{}, v8f{}, v8f{}};   // ctx 16x64
  float mrow[8], lrow[8];
#pragma unroll
  for (int r = 0; r < 8; ++r) { mrow[r] = -1e30f; lrow[r] = 0.f; }

  const float scale = 0.125f;                  // 1/sqrt(64)

  for (int kc = 0; kc < SS; kc += 64) {
    // ---- logits S = Q * K^T for 64 keys (four 16-key subtiles j)
    v8f sfr[4] = {v8f{}, v8f{}, v8f{}, v8f{}};
    const __bf16* kp0 = kT + ((size_t)bh * DHH + 0  + kb) * SS + kc + (lane & 15);
    const __bf16* kp1 = kT + ((size_t)bh * DHH + 32 + kb) * SS + kc + (lane & 15);
    __builtin_prefetch(kp0 + 64, 0, 3);        // next key chunk (K^T rows), WGP-near
    __builtin_prefetch(kp1 + 64, 0, 3);
#pragma unroll
    for (int j = 0; j < 4; ++j) {
      v16bf b0, b1;
#pragma unroll
      for (int s = 0; s < 16; ++s) {
        b0[s] = kp0[(size_t)(s + (s >> 3) * 8) * SS + j * 16];
        b1[s] = kp1[(size_t)(s + (s >> 3) * 8) * SS + j * 16];
      }
      sfr[j] = wmma_bf16(aq0, b0, sfr[j]);
      sfr[j] = wmma_bf16(aq1, b1, sfr[j]);
    }
    // ---- online softmax (row r lives in reg r, lanes within one 16-half)
    float alpha[8];
#pragma unroll
    for (int j = 0; j < 4; ++j)
#pragma unroll
      for (int r = 0; r < 8; ++r) sfr[j][r] *= scale;
#pragma unroll
    for (int r = 0; r < 8; ++r) {
      float t = fmaxf(fmaxf(sfr[0][r], sfr[1][r]), fmaxf(sfr[2][r], sfr[3][r]));
      t = fmaxf(t, __shfl_xor(t, 1));
      t = fmaxf(t, __shfl_xor(t, 2));
      t = fmaxf(t, __shfl_xor(t, 4));
      t = fmaxf(t, __shfl_xor(t, 8));
      float mnew = fmaxf(mrow[r], t);
      alpha[r] = __expf(mrow[r] - mnew);
      mrow[r] = mnew;
    }
    float rsum[8];
#pragma unroll
    for (int r = 0; r < 8; ++r) rsum[r] = 0.f;
#pragma unroll
    for (int j = 0; j < 4; ++j)
#pragma unroll
      for (int r = 0; r < 8; ++r) {
        float p = __expf(sfr[j][r] - mrow[r]);
        sfr[j][r] = p;
        rsum[r] += p;
      }
#pragma unroll
    for (int r = 0; r < 8; ++r) {
      float t = rsum[r];
      t += __shfl_xor(t, 1);
      t += __shfl_xor(t, 2);
      t += __shfl_xor(t, 4);
      t += __shfl_xor(t, 8);
      lrow[r] = lrow[r] * alpha[r] + t;
    }
    // ---- rescale running context
#pragma unroll
    for (int f = 0; f < 4; ++f)
#pragma unroll
      for (int r = 0; r < 8; ++r) acc[f][r] *= alpha[r];

    // ---- stage P (D-layout) -> LDS -> reload as A-fragments (intra-wave, DS in-order)
#pragma unroll
    for (int j = 0; j < 4; ++j)
#pragma unroll
      for (int r = 0; r < 8; ++r) {
        int pr = r + ((lane >> 4) << 3);
        int pc = j * 16 + (lane & 15);
        plds[wave][pr * 64 + pc] = f32_to_bf16(sfr[j][r]);
      }
    v16bf ap0 = load_afrag(&plds[wave][row * 64 + kb]);
    v16bf ap1 = load_afrag(&plds[wave][row * 64 + 32 + kb]);

    // ---- ctx += P * V  (lane = d column of B, 16 key rows gathered in A k-order)
    const __bf16* vp0 = v + ((size_t)bh * SS + kc + kb) * DHH + (lane & 15);
    const __bf16* vp1 = v + ((size_t)bh * SS + kc + 32 + kb) * DHH + (lane & 15);
    __builtin_prefetch(vp0 + 64 * DHH, 0, 3);  // next key chunk of V, WGP-near
#pragma unroll
    for (int f = 0; f < 4; ++f) {
      v16bf bv0, bv1;
#pragma unroll
      for (int s = 0; s < 16; ++s) {
        bv0[s] = vp0[(size_t)(s + (s >> 3) * 8) * DHH + f * 16];
        bv1[s] = vp1[(size_t)(s + (s >> 3) * 8) * DHH + f * 16];
      }
      acc[f] = wmma_bf16(ap0, bv0, acc[f]);
      acc[f] = wmma_bf16(ap1, bv1, acc[f]);
    }
  }

  // ---- finalize: ctx /= l, store to [B,S,H*DH] bf16
  const int b = bh >> 4, h = bh & 15;
  float invl[8];
#pragma unroll
  for (int r = 0; r < 8; ++r) invl[r] = 1.0f / lrow[r];
#pragma unroll
  for (int f = 0; f < 4; ++f) {
    int dcol = f * 16 + (lane & 15);
#pragma unroll
    for (int r = 0; r < 8; ++r) {
      int s = q0 + r + ((lane >> 4) << 3);
      ctxo[((size_t)(b * SS + s)) * DD + h * DHH + dcol] = f32_to_bf16(acc[f][r] * invl[r]);
    }
  }
}

// -------------------------------------------------------- output proj GEMM
__global__ void out_gemm(const __bf16* __restrict__ cb, const __bf16* __restrict__ W,
                         const float* __restrict__ bias, float* __restrict__ out) {
  const int lane = threadIdx.x & 31;
  const int wave = threadIdx.x >> 5;
  const int m0 = blockIdx.x * 128 + wave * 32;
  const int n0 = blockIdx.y * 64;
  const int row = lane & 15;
  const int kb  = (lane >> 4) * 8;

  v8f acc[2][4] = {{v8f{}, v8f{}, v8f{}, v8f{}}, {v8f{}, v8f{}, v8f{}, v8f{}}};

  for (int k0 = 0; k0 < DD; k0 += 32) {
    v16bf a0 = load_afrag(cb + (size_t)(m0 + row) * DD + k0 + kb);
    v16bf a1 = load_afrag(cb + (size_t)(m0 + 16 + row) * DD + k0 + kb);
    const __bf16* bp = W + (size_t)(k0 + kb) * DD + n0 + (lane & 15);
    __builtin_prefetch(bp + 32 * DD, 0, 3);
#pragma unroll
    for (int f = 0; f < 4; ++f) {
      v16bf bfrag;
#pragma unroll
      for (int s = 0; s < 16; ++s) bfrag[s] = bp[(size_t)(s + (s >> 3) * 8) * DD + f * 16];
      acc[0][f] = wmma_bf16(a0, bfrag, acc[0][f]);
      acc[1][f] = wmma_bf16(a1, bfrag, acc[1][f]);
    }
  }

#pragma unroll
  for (int t = 0; t < 2; ++t) {
#pragma unroll
    for (int f = 0; f < 4; ++f) {
      int n = n0 + f * 16 + (lane & 15);
      float bia = bias[n];
#pragma unroll
      for (int r = 0; r < 8; ++r) {
        int m = m0 + t * 16 + r + ((lane >> 4) << 3);
        out[(size_t)m * DD + n] = acc[t][f][r] + bia;
      }
    }
  }
}

extern "C" void kernel_launch(void* const* d_in, const int* in_sizes, int n_in,
                              void* d_out, int out_size, void* d_ws, size_t ws_size,
                              hipStream_t stream) {
  (void)in_sizes; (void)n_in; (void)out_size; (void)ws_size;
  const float* x  = (const float*)d_in[0];
  const float* Wq = (const float*)d_in[1];
  const float* bq = (const float*)d_in[2];
  const float* Wk = (const float*)d_in[3];
  const float* bk = (const float*)d_in[4];
  const float* Wv = (const float*)d_in[5];
  const float* bv = (const float*)d_in[6];
  const float* Wo = (const float*)d_in[7];
  const float* bo = (const float*)d_in[8];
  float* out = (float*)d_out;

  __bf16* xb = (__bf16*)d_ws;
  __bf16* wb = xb + (size_t)NX;          // 4 weight matrices, bf16
  __bf16* qw = wb + (size_t)4 * NW;      // Q  [B,H,S,DH]
  __bf16* kw = qw + (size_t)NX;          // K^T[B,H,DH,S]
  __bf16* vw = kw + (size_t)NX;          // V  [B,H,S,DH]
  __bf16* cw = vw + (size_t)NX;          // ctx bf16 [B,S,D]

  cvt_f32_bf16<<<NX / 1024, 256, 0, stream>>>(x,  xb, NX);
  cvt_f32_bf16<<<NW / 1024, 256, 0, stream>>>(Wq, wb + 0 * (size_t)NW, NW);
  cvt_f32_bf16<<<NW / 1024, 256, 0, stream>>>(Wk, wb + 1 * (size_t)NW, NW);
  cvt_f32_bf16<<<NW / 1024, 256, 0, stream>>>(Wv, wb + 2 * (size_t)NW, NW);
  cvt_f32_bf16<<<NW / 1024, 256, 0, stream>>>(Wo, wb + 3 * (size_t)NW, NW);

  qkv_gemm<<<dim3(32, 16, 3), 128, 0, stream>>>(xb, wb, bq, bk, bv, qw, kw, vw);
  attn_kernel<<<dim3(SS / 64, BB * HH), 128, 0, stream>>>(qw, kw, vw, cw);
  out_gemm<<<dim3(32, 16), 128, 0, stream>>>(cw, wb + 3 * (size_t)NW, bo, out);
}
